// BertSelfAttention_29059748725235
// MI455X (gfx1250) — matrix operand order repaired and checked by reading, not compiled, not run on hardware
//
#include <hip/hip_runtime.h>

typedef __bf16 bf16_t;
typedef __attribute__((ext_vector_type(8)))  __bf16 v8bf;
typedef __attribute__((ext_vector_type(16))) __bf16 v16bf;
typedef __attribute__((ext_vector_type(8)))  float  v8f;
typedef __attribute__((ext_vector_type(4)))  unsigned int u32x4;
typedef __attribute__((ext_vector_type(4)))  int  i32x4;
typedef __attribute__((ext_vector_type(8)))  int  i32x8;

constexpr int S  = 2048;
constexpr int H  = 1024;
constexpr int HD = 64;     // head dim
constexpr int NH = 16;     // heads

// ---------------------------------------------------------------------------
// WMMA helpers (CDNA5 16x16x32 bf16, f32 accumulate)
// ---------------------------------------------------------------------------
__device__ __forceinline__ v8f wmma_bf16(v16bf a, v16bf b, v8f c) {
  return __builtin_amdgcn_wmma_f32_16x16x32_bf16(
      /*neg_a=*/false, a, /*neg_b=*/false, b,
      /*c_mod=*/(short)0, c, /*reuse_a=*/false, /*reuse_b=*/false);
}

// A fragment: 16(M) x 32(K) bf16, row-major source with leading dim ld.
// Per ISA: lane<16 -> M=lane, K = {0..7, 16..23}; lane>=16 -> K = {8..15, 24..31}.
__device__ __forceinline__ v16bf load_a16(const bf16_t* base, int ld) {
  const int lane = threadIdx.x & 31;
  const int hf   = lane >> 4;
  const int m    = lane & 15;
  const bf16_t* p = base + (size_t)m * ld + hf * 8;
  v8bf lo = *(const v8bf*)(p);        // K = hf*8 + 0..7
  v8bf hi = *(const v8bf*)(p + 16);   // K = 16 + hf*8 + 0..7
  return __builtin_shufflevector(lo, hi, 0,1,2,3,4,5,6,7,8,9,10,11,12,13,14,15);
}

// B fragment: 32(K) x 16(N) bf16 from transposed-operand rows (rows = N).
// lane<16 -> N=lane, K = {0..7, 16..23}; lane>=16 -> N=lane-16, K = {8..15, 24..31}.
__device__ __forceinline__ v16bf load_b16(const bf16_t* baseT, int ld) {
  const int lane = threadIdx.x & 31;
  const int hf   = lane >> 4;
  const int n    = lane & 15;
  const bf16_t* p = baseT + (size_t)n * ld + hf * 8;
  v8bf lo = *(const v8bf*)(p);
  v8bf hi = *(const v8bf*)(p + 16);
  return __builtin_shufflevector(lo, hi, 0,1,2,3,4,5,6,7,8,9,10,11,12,13,14,15);
}

// ---------------------------------------------------------------------------
// TDM: async 2-D tile load Global -> LDS (bf16 elements).
// D# built per CDNA5 ISA §8.3/8.4: group0 = {count/type, lds_addr, gaddr},
// group1 = {data_size, tensor dims, tile dims, dim0 stride}.
// ---------------------------------------------------------------------------
__device__ __forceinline__ void tdm_load_2d(unsigned int lds_addr,
                                            const bf16_t* gptr,
                                            int tile_x, int tile_y,
                                            int stride_elems) {
  const unsigned long long ga = (unsigned long long)(uintptr_t)gptr;
  u32x4 g0;
  g0[0] = 1u;                                   // count=1, user mode, no gather
  g0[1] = lds_addr;                             // LDS byte address
  g0[2] = (unsigned int)ga;                     // global_addr[31:0]
  g0[3] = (unsigned int)((ga >> 32) & 0x01ffffffu) | (2u << 30);  // addr[56:32], type=2

  const unsigned int td0 = (unsigned int)stride_elems;  // tensor_dim0
  const unsigned int td1 = (1u << 20);                  // huge tensor_dim1 (no clip)
  const unsigned int st0 = (unsigned int)stride_elems;  // tensor_dim0_stride
  i32x8 g1;
  g1[0] = (int)(1u << 16);                                  // data_size = 2 bytes
  g1[1] = (int)((td0 & 0xffffu) << 16);                     // td0[15:0] @ [31:16]
  g1[2] = (int)(((td0 >> 16) & 0xffffu) | ((td1 & 0xffffu) << 16));
  g1[3] = (int)(((td1 >> 16) & 0xffffu) | (((unsigned)tile_x & 0xffffu) << 16));
  g1[4] = (int)((unsigned)tile_y & 0xffffu);                // tile_dim1; tile_dim2=0
  g1[5] = (int)st0;                                         // stride0[31:0]
  g1[6] = 0;                                                // stride0[47:32] | stride1 lo
  g1[7] = 0;

  i32x4 z4 = {0, 0, 0, 0};
#if __clang_major__ >= 23
  i32x8 z8 = {0, 0, 0, 0, 0, 0, 0, 0};
  __builtin_amdgcn_tensor_load_to_lds(g0, g1, z4, z4, z8, 0);
#else
  __builtin_amdgcn_tensor_load_to_lds(g0, g1, z4, z4, 0);
#endif
}

// ---------------------------------------------------------------------------
// Conversion kernels
// ---------------------------------------------------------------------------
__global__ void k_cast_bf16(const float* __restrict__ src,
                            bf16_t* __restrict__ dst, int n) {
  int i = blockIdx.x * blockDim.x + threadIdx.x;
  if (i < n) dst[i] = (bf16_t)src[i];
}

// W [H][H] row-major -> Wt with Wt[n][k] = W[k][n], cast to bf16
__global__ void k_cast_transpose(const float* __restrict__ W,
                                 bf16_t* __restrict__ Wt) {
  int i = blockIdx.x * blockDim.x + threadIdx.x;   // i < H*H
  int k = i >> 10;
  int n = i & (H - 1);
  Wt[(size_t)n * H + k] = (bf16_t)W[i];
}

// ---------------------------------------------------------------------------
// QKV projection: [S,H] x [H,H] + bias -> Q [S,H], K [S,H], Vt [H,S] (bf16)
// grid (S/64, H/64, 3), block 128 (4 waves, each a 16x64 tile).
// Manual 2x ping-pong unroll: two independent fragment sets, reloaded (never
// copied), so loads of one phase overlap the WMMA chain of the other.
// ---------------------------------------------------------------------------
__global__ __launch_bounds__(128) void k_qkv_gemm(
    const bf16_t* __restrict__ X,
    const bf16_t* __restrict__ WqT, const bf16_t* __restrict__ WkT,
    const bf16_t* __restrict__ WvT,
    const float* __restrict__ bq, const float* __restrict__ bk,
    const float* __restrict__ bv,
    bf16_t* __restrict__ Q, bf16_t* __restrict__ K, bf16_t* __restrict__ Vt) {
  const int z    = blockIdx.z;
  const bf16_t* WT  = (z == 0) ? WqT : (z == 1) ? WkT : WvT;
  const float*  bia = (z == 0) ? bq  : (z == 1) ? bk  : bv;
  const int wave = threadIdx.x >> 5;
  const int lane = threadIdx.x & 31;
  const int hf   = lane >> 4;
  const int col  = lane & 15;
  const int m0   = blockIdx.x * 64 + wave * 16;
  const int n0   = blockIdx.y * 64;

  v8f acc[4];
#pragma unroll
  for (int t = 0; t < 4; ++t) acc[t] = (v8f){0,0,0,0,0,0,0,0};

  const bf16_t* Xrow = X + (size_t)m0 * H;
  const bf16_t* Wrow = WT + (size_t)n0 * H;

  v16bf aP = load_a16(Xrow, H);
  v16bf bP[4];
#pragma unroll
  for (int t = 0; t < 4; ++t) bP[t] = load_b16(Wrow + (size_t)(t * 16) * H, H);

  for (int k = 0; k < H; k += 64) {
    __builtin_prefetch((const void*)(Wrow + k + 1024), 0, 1);
    // pong fragments for k+32 (always in range: H % 64 == 0)
    v16bf aQ = load_a16(Xrow + k + 32, H);
    v16bf bQ[4];
#pragma unroll
    for (int t = 0; t < 4; ++t)
      bQ[t] = load_b16(Wrow + (size_t)(t * 16) * H + k + 32, H);

#pragma unroll
    for (int t = 0; t < 4; ++t) acc[t] = wmma_bf16(aP, bP[t], acc[t]);

    if (k + 64 < H) {   // ping fragments for k+64
      aP = load_a16(Xrow + k + 64, H);
#pragma unroll
      for (int t = 0; t < 4; ++t)
        bP[t] = load_b16(Wrow + (size_t)(t * 16) * H + k + 64, H);
    }
#pragma unroll
    for (int t = 0; t < 4; ++t) acc[t] = wmma_bf16(aQ, bQ[t], acc[t]);
  }

#pragma unroll
  for (int t = 0; t < 4; ++t) {
    const int c  = n0 + t * 16 + col;
    const float bvv = bia[c];
#pragma unroll
    for (int i = 0; i < 8; ++i) {
      const int row = m0 + i + 8 * hf;
      const float v = acc[t][i] + bvv;
      if (z == 0)      Q[(size_t)row * H + c] = (bf16_t)v;
      else if (z == 1) K[(size_t)row * H + c] = (bf16_t)v;
      else             Vt[(size_t)c * S + row] = (bf16_t)v;  // transposed V
    }
  }
}

// ---------------------------------------------------------------------------
// Flash attention: grid (S/16, NH), block 32 (one wave = 16 query rows).
// K/V tiles stream through LDS via double-buffered TDM (tensor_load_to_lds),
// synchronized with s_wait_tensorcnt.  P staged in LDS for the PV A-fragments.
// ---------------------------------------------------------------------------
__global__ __launch_bounds__(32) void k_attn(
    const bf16_t* __restrict__ Q, const bf16_t* __restrict__ Kb,
    const bf16_t* __restrict__ Vt, bf16_t* __restrict__ ctx) {
  const int h    = blockIdx.y;
  const int lane = threadIdx.x & 31;
  const int hf   = lane >> 4;
  const int col  = lane & 15;
  const int q0   = blockIdx.x * 16;

  __shared__ bf16_t Ksh[2][64 * 64];   // [key][d] per buffer
  __shared__ bf16_t Vsh[2][64 * 64];   // [d][key] per buffer
  __shared__ bf16_t Psh[16 * 64];

  const unsigned int k_lds0 = (unsigned int)(uintptr_t)(void*)&Ksh[0][0];
  const unsigned int k_lds1 = (unsigned int)(uintptr_t)(void*)&Ksh[1][0];
  const unsigned int v_lds0 = (unsigned int)(uintptr_t)(void*)&Vsh[0][0];
  const unsigned int v_lds1 = (unsigned int)(uintptr_t)(void*)&Vsh[1][0];

  // Q fragments for this wave's 16 rows (K-dim = head dim 64 -> 2 chunks)
  const bf16_t* qbase = Q + (size_t)q0 * H + h * HD;
  v16bf qa0 = load_a16(qbase + 0,  H);
  v16bf qa1 = load_a16(qbase + 32, H);

  float m_i[8], l_i[8];
  v8f   cacc[4];
#pragma unroll
  for (int i = 0; i < 8; ++i) { m_i[i] = -3.0e38f; l_i[i] = 0.0f; }
#pragma unroll
  for (int t = 0; t < 4; ++t) cacc[t] = (v8f){0,0,0,0,0,0,0,0};

  // Prologue: TDM-load the first K/V tiles into buffer 0.
  tdm_load_2d(k_lds0, Kb + (size_t)0 * H + h * HD, 64, 64, H);
  tdm_load_2d(v_lds0, Vt + (size_t)(h * HD) * S + 0, 64, 64, S);

  for (int kb = 0; kb < S; kb += 64) {
    const int cur = (kb >> 6) & 1;
    // Kick off next tiles, then wait only for the current buffer's 2 TDM ops
    if (kb + 64 < S) {
      const unsigned int kl = cur ? k_lds0 : k_lds1;
      const unsigned int vl = cur ? v_lds0 : v_lds1;
      tdm_load_2d(kl, Kb + (size_t)(kb + 64) * H + h * HD, 64, 64, H);
      tdm_load_2d(vl, Vt + (size_t)(h * HD) * S + (kb + 64), 64, 64, S);
      __builtin_amdgcn_s_wait_tensorcnt(2);
    } else {
      __builtin_amdgcn_s_wait_tensorcnt(0);
    }
    asm volatile("" ::: "memory");
    const bf16_t* Kl = Ksh[cur];
    const bf16_t* Vl = Vsh[cur];

    // ---- preload all 8 K-tile B fragments, then the QK^T WMMA chain ----
    v16bf kf0[4], kf1[4];
#pragma unroll
    for (int t = 0; t < 4; ++t) {
      kf0[t] = load_b16(Kl + (t * 16) * 64 + 0,  64);
      kf1[t] = load_b16(Kl + (t * 16) * 64 + 32, 64);
    }
    v8f s[4];
#pragma unroll
    for (int t = 0; t < 4; ++t) {
      s[t] = (v8f){0,0,0,0,0,0,0,0};
      s[t] = wmma_bf16(qa0, kf0[t], s[t]);
      s[t] = wmma_bf16(qa1, kf1[t], s[t]);
    }
    const float scale = 0.125f;  // 1/sqrt(64)
#pragma unroll
    for (int t = 0; t < 4; ++t)
#pragma unroll
      for (int i = 0; i < 8; ++i) s[t][i] *= scale;

    // ---- online softmax over this 64-key block ----
    float bm[8];
#pragma unroll
    for (int i = 0; i < 8; ++i)
      bm[i] = fmaxf(fmaxf(s[0][i], s[1][i]), fmaxf(s[2][i], s[3][i]));
#pragma unroll
    for (int off = 1; off < 16; off <<= 1)
#pragma unroll
      for (int i = 0; i < 8; ++i)
        bm[i] = fmaxf(bm[i], __shfl_xor(bm[i], off, 32));

    float corr[8], rs[8];
#pragma unroll
    for (int i = 0; i < 8; ++i) {
      float mn = fmaxf(m_i[i], bm[i]);
      corr[i] = __expf(m_i[i] - mn);
      m_i[i]  = mn;
      rs[i]   = 0.0f;
    }
#pragma unroll
    for (int t = 0; t < 4; ++t)
#pragma unroll
      for (int i = 0; i < 8; ++i) {
        float p = __expf(s[t][i] - m_i[i]);
        s[t][i] = p;
        rs[i] += p;
      }
#pragma unroll
    for (int off = 1; off < 16; off <<= 1)
#pragma unroll
      for (int i = 0; i < 8; ++i) rs[i] += __shfl_xor(rs[i], off, 32);
#pragma unroll
    for (int i = 0; i < 8; ++i) l_i[i] = l_i[i] * corr[i] + rs[i];
#pragma unroll
    for (int t = 0; t < 4; ++t)
#pragma unroll
      for (int i = 0; i < 8; ++i) cacc[t][i] *= corr[i];

    // ---- P (C-layout) -> LDS (row-major 16x64 bf16) ----
#pragma unroll
    for (int t = 0; t < 4; ++t)
#pragma unroll
      for (int i = 0; i < 8; ++i)
        Psh[(i + 8 * hf) * 64 + t * 16 + col] = (bf16_t)s[t][i];
    asm volatile("s_wait_dscnt 0" ::: "memory");   // wave-local LDS RAW

    // ---- preload V fragments + P A-fragments, then the PV WMMA chain ----
    v16bf vf0[4], vf1[4];
#pragma unroll
    for (int t = 0; t < 4; ++t) {
      vf0[t] = load_b16(Vl + (t * 16) * 64 + 0,  64);
      vf1[t] = load_b16(Vl + (t * 16) * 64 + 32, 64);
    }
    v16bf pa0 = load_a16(Psh + 0,  64);
    v16bf pa1 = load_a16(Psh + 32, 64);
#pragma unroll
    for (int t = 0; t < 4; ++t) {
      cacc[t] = wmma_bf16(pa0, vf0[t], cacc[t]);
      cacc[t] = wmma_bf16(pa1, vf1[t], cacc[t]);
    }
  }

  // ---- normalize and store ctx as bf16 [S, H] ----
  float inv[8];
#pragma unroll
  for (int i = 0; i < 8; ++i) inv[i] = 1.0f / l_i[i];
#pragma unroll
  for (int t = 0; t < 4; ++t)
#pragma unroll
    for (int i = 0; i < 8; ++i) {
      const int row = q0 + i + 8 * hf;
      const int c   = h * HD + t * 16 + col;
      ctx[(size_t)row * H + c] = (bf16_t)(cacc[t][i] * inv[i]);
    }
}

// ---------------------------------------------------------------------------
// Output projection: ctx [S,H] bf16 x WoT + bo -> out [S,H] fp32
// grid (S/64, H/64), block 128; same ping-pong pipeline as k_qkv_gemm.
// ---------------------------------------------------------------------------
__global__ __launch_bounds__(128) void k_out_gemm(
    const bf16_t* __restrict__ Xc, const bf16_t* __restrict__ WoT,
    const float* __restrict__ bo, float* __restrict__ out) {
  const int wave = threadIdx.x >> 5;
  const int lane = threadIdx.x & 31;
  const int hf   = lane >> 4;
  const int col  = lane & 15;
  const int m0   = blockIdx.x * 64 + wave * 16;
  const int n0   = blockIdx.y * 64;

  v8f acc[4];
#pragma unroll
  for (int t = 0; t < 4; ++t) acc[t] = (v8f){0,0,0,0,0,0,0,0};

  const bf16_t* Xrow = Xc + (size_t)m0 * H;
  const bf16_t* Wrow = WoT + (size_t)n0 * H;

  v16bf aP = load_a16(Xrow, H);
  v16bf bP[4];
#pragma unroll
  for (int t = 0; t < 4; ++t) bP[t] = load_b16(Wrow + (size_t)(t * 16) * H, H);

  for (int k = 0; k < H; k += 64) {
    __builtin_prefetch((const void*)(Wrow + k + 1024), 0, 1);
    v16bf aQ = load_a16(Xrow + k + 32, H);
    v16bf bQ[4];
#pragma unroll
    for (int t = 0; t < 4; ++t)
      bQ[t] = load_b16(Wrow + (size_t)(t * 16) * H + k + 32, H);

#pragma unroll
    for (int t = 0; t < 4; ++t) acc[t] = wmma_bf16(aP, bP[t], acc[t]);

    if (k + 64 < H) {
      aP = load_a16(Xrow + k + 64, H);
#pragma unroll
      for (int t = 0; t < 4; ++t)
        bP[t] = load_b16(Wrow + (size_t)(t * 16) * H + k + 64, H);
    }
#pragma unroll
    for (int t = 0; t < 4; ++t) acc[t] = wmma_bf16(aQ, bQ[t], acc[t]);
  }

#pragma unroll
  for (int t = 0; t < 4; ++t) {
    const int c = n0 + t * 16 + col;
    const float bvv = bo[c];
#pragma unroll
    for (int i = 0; i < 8; ++i) {
      const int row = m0 + i + 8 * hf;
      out[(size_t)row * H + c] = acc[t][i] + bvv;
    }
  }
}

// ---------------------------------------------------------------------------
extern "C" void kernel_launch(void* const* d_in, const int* in_sizes, int n_in,
                              void* d_out, int out_size, void* d_ws, size_t ws_size,
                              hipStream_t stream) {
  const float* x  = (const float*)d_in[0];
  const float* Wq = (const float*)d_in[1];
  const float* bq = (const float*)d_in[2];
  const float* Wk = (const float*)d_in[3];
  const float* bk = (const float*)d_in[4];
  const float* Wv = (const float*)d_in[5];
  const float* bv = (const float*)d_in[6];
  const float* Wo = (const float*)d_in[7];
  const float* bo = (const float*)d_in[8];
  float* out = (float*)d_out;

  // workspace partition (all bf16): 28 MB total
  bf16_t* p    = (bf16_t*)d_ws;
  bf16_t* xbf  = p; p += (size_t)S * H;
  bf16_t* WqT  = p; p += (size_t)H * H;
  bf16_t* WkT  = p; p += (size_t)H * H;
  bf16_t* WvT  = p; p += (size_t)H * H;
  bf16_t* WoT  = p; p += (size_t)H * H;
  bf16_t* Qb   = p; p += (size_t)S * H;
  bf16_t* Kb   = p; p += (size_t)S * H;
  bf16_t* Vt   = p; p += (size_t)S * H;
  bf16_t* ctxb = p; p += (size_t)S * H;

  const int nSH = S * H;
  const int nHH = H * H;
  k_cast_bf16<<<(nSH + 255) / 256, 256, 0, stream>>>(x, xbf, nSH);
  k_cast_transpose<<<nHH / 256, 256, 0, stream>>>(Wq, WqT);
  k_cast_transpose<<<nHH / 256, 256, 0, stream>>>(Wk, WkT);
  k_cast_transpose<<<nHH / 256, 256, 0, stream>>>(Wv, WvT);
  k_cast_transpose<<<nHH / 256, 256, 0, stream>>>(Wo, WoT);

  k_qkv_gemm<<<dim3(S / 64, H / 64, 3), 128, 0, stream>>>(
      xbf, WqT, WkT, WvT, bq, bk, bv, Qb, Kb, Vt);

  k_attn<<<dim3(S / 16, NH), 32, 0, stream>>>(Qb, Kb, Vt, ctxb);

  k_out_gemm<<<dim3(S / 64, H / 64), 128, 0, stream>>>(ctxb, WoT, bo, out);
}
